// ReformerHead_18683107737675
// MI455X (gfx1250) — matrix operand order repaired and checked by reading, not compiled, not run on hardware
//
#include <hip/hip_runtime.h>
#include <hip/hip_bf16.h>
#include <math.h>

// ---------------- problem constants ----------------
static constexpr int DIMc    = 256;
static constexpr int HEADS   = 8;
static constexpr int DH      = 32;
static constexpr int Tc      = 4096;   // NVEC
static constexpr int Bc      = 4;
static constexpr int NHASH   = 4;
static constexpr int BUCKET  = 64;
static constexpr int NBUCK   = 64;     // NVEC/BUCKET
static constexpr int BH      = Bc * HEADS;      // 32
static constexpr int Rc      = Bc * Tc;         // 16384 rows
static constexpr int Sc      = NHASH * Tc;      // 16384 sorted positions per bh
static constexpr int CHUNKS  = Sc / BUCKET;     // 256
static constexpr int FF      = 4 * DIMc;        // 1024
static constexpr int NCLASS  = 2;
static constexpr float MASK_NEG = -1.0e9f;
static constexpr float SELF_VAL = -50000.0f;

typedef __attribute__((ext_vector_type(16))) __bf16 v16bf;
typedef __attribute__((ext_vector_type(8)))  __bf16 v8bf;
typedef __attribute__((ext_vector_type(8)))  float  v8f;

__device__ __forceinline__ float gelu_f(float x) {
    // jax.nn.gelu default (tanh approximation)
    float x3 = x * x * x;
    return 0.5f * x * (1.0f + tanhf(0.7978845608028654f * (x + 0.044715f * x3)));
}

// ---------------- weight prep: f32 (K x N) -> bf16 (N x K) ----------------
__global__ void transpose_cast_kernel(const float* __restrict__ src,
                                      __bf16* __restrict__ dst, int K, int N) {
    int idx = blockIdx.x * 256 + threadIdx.x;
    if (idx >= K * N) return;
    int k = idx / N, n = idx % N;
    dst[(size_t)n * K + k] = (__bf16)src[idx];
}

// ---------------- embedding + posenc -> x1, x2 ----------------
__global__ void embed_kernel(const int* __restrict__ ids, const float* __restrict__ emb,
                             const float* __restrict__ pos,
                             float* __restrict__ x1, float* __restrict__ x2) {
    int row = blockIdx.x;            // 0..R-1
    int k   = threadIdx.x;           // 0..255
    int id  = ids[row];
    int t   = row % Tc;
    float h = emb[(size_t)id * DIMc + k] + pos[(size_t)t * DIMc + k];
    size_t o = (size_t)row * DIMc + k;
    x1[o] = h; x2[o] = h;
}

// ---------------- layernorm (one wave per row) -> bf16 ----------------
__global__ void ln_kernel(const float* __restrict__ x, const float* __restrict__ s,
                          const float* __restrict__ b_, __bf16* __restrict__ out) {
    int row  = blockIdx.x;
    int lane = threadIdx.x;          // 32 lanes
    const float* xp = x + (size_t)row * DIMc;
    float v[8], sum = 0.f, sq = 0.f;
#pragma unroll
    for (int i = 0; i < 8; i++) { v[i] = xp[lane + i * 32]; sum += v[i]; sq += v[i] * v[i]; }
#pragma unroll
    for (int m = 16; m > 0; m >>= 1) { sum += __shfl_xor(sum, m, 32); sq += __shfl_xor(sq, m, 32); }
    float mean = sum * (1.0f / DIMc);
    float var  = sq * (1.0f / DIMc) - mean * mean;
    float inv  = rsqrtf(var + 1e-5f);
#pragma unroll
    for (int i = 0; i < 8; i++) {
        int col = lane + i * 32;
        out[(size_t)row * DIMc + col] = (__bf16)((v[i] - mean) * inv * s[col] + b_[col]);
    }
}

// ---------------- generic WMMA GEMM: out = A(RxK,bf16) @ Bt^T(NxK,bf16) ----------------
// Each wave computes a 16x64 output tile (4 accumulators): A fragment loaded once
// per k-step and reused across 4 B fragments -> 4 wmma per 6 vector loads.
// Block = 4 waves = 16x256 of output; grid = (R/16, N/256).
template <bool BIAS, bool GELU_, bool RES, bool OUTF, bool OUTB>
__global__ void gemm_kernel(const __bf16* __restrict__ A, const __bf16* __restrict__ Bt,
                            const float* __restrict__ bias,
                            float* __restrict__ outf, __bf16* __restrict__ outb,
                            int K, int N) {
    int lane = threadIdx.x, half = lane >> 4, mn = lane & 15;
    int row0    = blockIdx.x * 16;
    int colbase = blockIdx.y * 256 + threadIdx.y * 64;
    v8f acc[4] = {};
    const __bf16* arow = A + (size_t)(row0 + mn) * K;
    for (int k0 = 0; k0 < K; k0 += 32) {
        v8bf alo = *(const v8bf*)(arow + k0 + half * 16);
        v8bf ahi = *(const v8bf*)(arow + k0 + half * 16 + 8);
        v16bf a;
#pragma unroll
        for (int i = 0; i < 8; i++) { a[i] = alo[i]; a[i + 8] = ahi[i]; }
#pragma unroll
        for (int nt = 0; nt < 4; nt++) {
            v16bf b = *(const v16bf*)(Bt + (size_t)(colbase + nt * 16 + mn) * K + k0 + half * 16);
            acc[nt] = __builtin_amdgcn_wmma_f32_16x16x32_bf16(false, a, false, b, (short)0,
                                                              acc[nt], false, false);
        }
    }
#pragma unroll
    for (int nt = 0; nt < 4; nt++) {
        int col = colbase + nt * 16 + mn;
        float bv = BIAS ? bias[col] : 0.f;
#pragma unroll
        for (int r = 0; r < 8; r++) {
            int row = row0 + r + 8 * half;
            float val = acc[nt][r] + bv;
            if (GELU_) val = gelu_f(val);
            size_t o = (size_t)row * N + col;
            if (RES)  val += outf[o];
            if (OUTF) outf[o] = val;
            if (OUTB) outb[o] = (__bf16)val;
        }
    }
}

// ---------------- LSH bucket assignment ----------------
__global__ void bucket_kernel(const float* __restrict__ qk, const float* __restrict__ rot,
                              int* __restrict__ buckets) {
    int idx = blockIdx.x * 256 + threadIdx.x;
    if (idx >= BH * NHASH * Tc) return;
    int bh  = idx / (NHASH * Tc);
    int rem = idx % (NHASH * Tc);
    int h   = rem / Tc;
    int t   = rem % Tc;
    int b = bh / HEADS, head = bh % HEADS;
    const float* q = qk + ((size_t)(b * Tc + t)) * DIMc + head * DH;
    float qv[DH];
#pragma unroll
    for (int d = 0; d < DH; d++) qv[d] = q[d];
    float best = -3.4e38f, bestn = -3.4e38f;
    int bi = 0, bni = 0;
    for (int r = 0; r < NBUCK / 2; r++) {
        float acc = 0.f;
#pragma unroll
        for (int d = 0; d < DH; d++) acc += qv[d] * rot[((size_t)d * NHASH + h) * (NBUCK / 2) + r];
        if (acc  > best)  { best  = acc;  bi  = r; }
        if (-acc > bestn) { bestn = -acc; bni = r; }
    }
    int am = (best >= bestn) ? bi : (NBUCK / 2 + bni);
    buckets[(size_t)bh * Sc + rem] = am + h * NBUCK;
}

// ---------------- stable counting sort by bucket (one block per bh) ----------------
__global__ void sort_kernel(const int* __restrict__ buckets, int* __restrict__ st,
                            int* __restrict__ undo) {
    int bh = blockIdx.x, b = threadIdx.x;    // 256 threads = 256 bucket ids
    __shared__ int cnt[256];
    __shared__ int offs[256];
    const int* bu = buckets + (size_t)bh * Sc;
    int c = 0;
    for (int j = 0; j < Sc; j++) c += (bu[j] == b) ? 1 : 0;
    cnt[b] = c;
    __syncthreads();
    if (b == 0) { int a = 0; for (int i = 0; i < 256; i++) { offs[i] = a; a += cnt[i]; } }
    __syncthreads();
    int pos = offs[b];
    int* stp = st   + (size_t)bh * Sc;
    int* up  = undo + (size_t)bh * Sc;
    for (int j = 0; j < Sc; j++) {
        if (bu[j] == b) { stp[pos] = j % Tc; up[j] = pos; pos++; }
    }
}

// ---------------- gather sorted q/k/v (k normalized) as bf16 ----------------
__global__ void gather_kernel(const float* __restrict__ qk, const float* __restrict__ v,
                              const int* __restrict__ st, __bf16* __restrict__ sqk,
                              __bf16* __restrict__ bk, __bf16* __restrict__ sv) {
    int idx = blockIdx.x * 256 + threadIdx.x;
    if (idx >= BH * Sc) return;
    int bh = idx / Sc;
    int t  = st[idx];
    int b = bh / HEADS, head = bh % HEADS;
    const float* qp = qk + ((size_t)(b * Tc + t)) * DIMc + head * DH;
    const float* vp = v  + ((size_t)(b * Tc + t)) * DIMc + head * DH;
    float qv[DH], nrm = 0.f;
#pragma unroll
    for (int d = 0; d < DH; d++) { qv[d] = qp[d]; nrm += qv[d] * qv[d]; }
    nrm = fmaxf(sqrtf(nrm), 1e-12f);
    float inv = 1.0f / nrm;
    size_t o = (size_t)idx * DH;
#pragma unroll
    for (int d = 0; d < DH; d++) {
        sqk[o + d] = (__bf16)qv[d];
        bk[o + d]  = (__bf16)(qv[d] * inv);
        sv[o + d]  = (__bf16)vp[d];
    }
}

// ---------------- chunked attention: dots + softmax + probs@V (WMMA) ----------------
__global__ void attn_kernel(const __bf16* __restrict__ sqk, const __bf16* __restrict__ bk,
                            const __bf16* __restrict__ sv, const int* __restrict__ st,
                            __bf16* __restrict__ so, float* __restrict__ slog) {
    __shared__ __bf16 lds_bk[128][32];
    __shared__ __bf16 lds_v[128][32];
    __shared__ int    lds_tk[128];
    __shared__ int    lds_tq[64];
    __shared__ float  lds_p[4][16][128];

    int bh = blockIdx.x / CHUNKS;
    int c  = blockIdx.x % CHUNKS;
    int cp = (c + CHUNKS - 1) % CHUNKS;   // look-one-back (roll +1)
    int tid = threadIdx.y * 32 + threadIdx.x;   // 0..127
    size_t base = (size_t)bh * Sc;

    { // stage keys/values/time indices
        int j   = tid;
        int src = (j < BUCKET) ? (c * BUCKET + j) : (cp * BUCKET + (j - BUCKET));
        size_t sp = base + src;
#pragma unroll
        for (int d = 0; d < DH; d++) { lds_bk[j][d] = bk[sp * DH + d]; lds_v[j][d] = sv[sp * DH + d]; }
        lds_tk[j] = st[sp];
        if (j < BUCKET) lds_tq[j] = st[base + c * BUCKET + j];
    }
    __syncthreads();

    int w = threadIdx.y, lane = threadIdx.x, half = lane >> 4, mn = lane & 15;
    int m0 = w * 16;

    // A fragment: 16 query rows (K = DH = 32)
    size_t qrow = (base + c * BUCKET + m0 + mn) * DH;
    v8bf qlo = *(const v8bf*)(sqk + qrow + half * 16);
    v8bf qhi = *(const v8bf*)(sqk + qrow + half * 16 + 8);
    v16bf aq;
#pragma unroll
    for (int i = 0; i < 8; i++) { aq[i] = qlo[i]; aq[i + 8] = qhi[i]; }

    const float scale = 0.17677669529663687f; // 32^-0.5
    for (int nt = 0; nt < 8; nt++) {
        int n = nt * 16 + mn;
        v16bf bb = *(const v16bf*)(&lds_bk[n][half * 16]);
        v8f d = {};
        d = __builtin_amdgcn_wmma_f32_16x16x32_bf16(false, aq, false, bb, (short)0, d, false, false);
#pragma unroll
        for (int r = 0; r < 8; r++) {
            int mrow = r + 8 * half;
            float val = d[r] * scale;
            int tq = lds_tq[m0 + mrow], tk = lds_tk[n];
            if (tq < tk)       val = MASK_NEG;
            else if (tq == tk) val = SELF_VAL;
            lds_p[w][mrow][n] = val;
        }
    }
    __syncthreads();

    // softmax per query row (lanes 0..15 of each wave handle one row each)
    if (lane < 16) {
        float* row = lds_p[w][lane];
        float mx = -3.4e38f;
        for (int jj = 0; jj < 128; jj++) mx = fmaxf(mx, row[jj]);
        float ssum = 0.f;
        for (int jj = 0; jj < 128; jj++) ssum += expf(row[jj] - mx);
        float lse = mx + logf(ssum);
        slog[base + c * BUCKET + m0 + lane] = lse;
        for (int jj = 0; jj < 128; jj++) row[jj] = expf(row[jj] - lse);
    }
    __syncthreads();

    // probs (16x128) @ V (128x32) via 2 N-tiles x 4 K-steps
    for (int nt = 0; nt < 2; nt++) {
        v8f o = {};
        for (int ks = 0; ks < 4; ks++) {
            v16bf ap;
#pragma unroll
            for (int j = 0; j < 8; j++) {
                ap[j]     = (__bf16)lds_p[w][mn][ks * 32 + half * 16 + j];
                ap[j + 8] = (__bf16)lds_p[w][mn][ks * 32 + half * 16 + 8 + j];
            }
            int n = nt * 16 + mn;
            v16bf bp;
#pragma unroll
            for (int j = 0; j < 16; j++) bp[j] = lds_v[ks * 32 + half * 16 + j][n];
            o = __builtin_amdgcn_wmma_f32_16x16x32_bf16(false, ap, false, bp, (short)0, o, false, false);
        }
#pragma unroll
        for (int r = 0; r < 8; r++) {
            int mrow = r + 8 * half;
            so[(base + c * BUCKET + m0 + mrow) * DH + nt * 16 + mn] = (__bf16)o[r];
        }
    }
}

// ---------------- unsort + combine hash rounds -> merged heads bf16 ----------------
__global__ void combine_kernel(const int* __restrict__ undo, const float* __restrict__ slog,
                               const __bf16* __restrict__ so, __bf16* __restrict__ attn) {
    int idx = blockIdx.x * 256 + threadIdx.x;
    if (idx >= BH * Tc) return;
    int bh = idx / Tc, t = idx % Tc;
    int b = bh / HEADS, head = bh % HEADS;
    size_t base = (size_t)bh * Sc;
    int pp[NHASH]; float lg[NHASH];
#pragma unroll
    for (int h = 0; h < NHASH; h++) { pp[h] = undo[base + h * Tc + t]; lg[h] = slog[base + pp[h]]; }
    float mx = lg[0];
#pragma unroll
    for (int h = 1; h < NHASH; h++) mx = fmaxf(mx, lg[h]);
    float wgt[NHASH], ssum = 0.f;
#pragma unroll
    for (int h = 0; h < NHASH; h++) { wgt[h] = expf(lg[h] - mx); ssum += wgt[h]; }
    float inv = 1.0f / ssum;
    __bf16* op = attn + ((size_t)(b * Tc + t)) * DIMc + head * DH;
    for (int d = 0; d < DH; d++) {
        float o = 0.f;
#pragma unroll
        for (int h = 0; h < NHASH; h++) o += wgt[h] * inv * (float)so[(base + pp[h]) * DH + d];
        op[d] = (__bf16)o;
    }
}

// ---------------- final classifier ----------------
__global__ void cls_kernel(const float* __restrict__ x1, const float* __restrict__ x2,
                           const float* __restrict__ wc, const float* __restrict__ bc,
                           float* __restrict__ out) {
    int tid = threadIdx.x;
    if (tid >= Bc * NCLASS) return;
    int b = tid / NCLASS, c = tid % NCLASS;
    const float* r1 = x1 + (size_t)(b * Tc) * DIMc;
    const float* r2 = x2 + (size_t)(b * Tc) * DIMc;
    float acc = 0.f;
    for (int k = 0; k < DIMc; k++) acc += 0.5f * (r1[k] + r2[k]) * wc[k * NCLASS + c];
    out[tid] = acc + bc[c];
}

// ---------------- driver ----------------
extern "C" void kernel_launch(void* const* d_in, const int* in_sizes, int n_in,
                              void* d_out, int out_size, void* d_ws, size_t ws_size,
                              hipStream_t stream) {
    (void)in_sizes; (void)n_in; (void)out_size; (void)ws_size;
    const int*   ids    = (const int*)  d_in[0];
    const float* emb    = (const float*)d_in[1];
    const float* posenc = (const float*)d_in[2];
    const float* ln1_s  = (const float*)d_in[3];
    const float* ln1_b  = (const float*)d_in[4];
    const float* w_qk   = (const float*)d_in[5];
    const float* w_v    = (const float*)d_in[6];
    const float* w_out  = (const float*)d_in[7];
    const float* b_out  = (const float*)d_in[8];
    const float* ln2_s  = (const float*)d_in[9];
    const float* ln2_b  = (const float*)d_in[10];
    const float* w_ff1  = (const float*)d_in[11];
    const float* b_ff1  = (const float*)d_in[12];
    const float* w_ff2  = (const float*)d_in[13];
    const float* b_ff2  = (const float*)d_in[14];
    const float* rot    = (const float*)d_in[15];
    const float* w_cls  = (const float*)d_in[16];
    const float* b_cls  = (const float*)d_in[17];
    float* out = (float*)d_out;

    char* wsp = (char*)d_ws;
    auto alloc = [&](size_t bytes) -> char* {
        char* p = wsp; wsp += (bytes + 255) & ~(size_t)255; return p;
    };
    float*  x1    = (float*) alloc((size_t)Rc * DIMc * 4);
    float*  x2    = (float*) alloc((size_t)Rc * DIMc * 4);
    __bf16* xn    = (__bf16*)alloc((size_t)Rc * DIMc * 2);
    float*  qk    = (float*) alloc((size_t)Rc * DIMc * 4);
    float*  vv    = (float*) alloc((size_t)Rc * DIMc * 4);
    int*    bkt   = (int*)   alloc((size_t)BH * Sc * 4);
    int*    st    = (int*)   alloc((size_t)BH * Sc * 4);
    int*    undo  = (int*)   alloc((size_t)BH * Sc * 4);
    __bf16* sqk   = (__bf16*)alloc((size_t)BH * Sc * DH * 2);
    __bf16* bkn   = (__bf16*)alloc((size_t)BH * Sc * DH * 2);
    __bf16* svb   = (__bf16*)alloc((size_t)BH * Sc * DH * 2);
    __bf16* sob   = (__bf16*)alloc((size_t)BH * Sc * DH * 2);
    float*  slog  = (float*) alloc((size_t)BH * Sc * 4);
    __bf16* attn  = (__bf16*)alloc((size_t)Rc * DIMc * 2);
    __bf16* hid   = (__bf16*)alloc((size_t)Rc * FF * 2);
    __bf16* wqkT  = (__bf16*)alloc((size_t)DIMc * DIMc * 2);
    __bf16* wvT   = (__bf16*)alloc((size_t)DIMc * DIMc * 2);
    __bf16* woT   = (__bf16*)alloc((size_t)DIMc * DIMc * 2);
    __bf16* wf1T  = (__bf16*)alloc((size_t)DIMc * FF * 2);
    __bf16* wf2T  = (__bf16*)alloc((size_t)DIMc * FF * 2);

    dim3 gblk(32, 4);

    embed_kernel<<<Rc, 256, 0, stream>>>(ids, emb, posenc, x1, x2);

    for (int l = 0; l < 2; l++) {
        const float* wq = w_qk  + (size_t)l * DIMc * DIMc;
        const float* wv = w_v   + (size_t)l * DIMc * DIMc;
        const float* wo = w_out + (size_t)l * DIMc * DIMc;
        const float* w1 = w_ff1 + (size_t)l * DIMc * FF;
        const float* w2 = w_ff2 + (size_t)l * FF * DIMc;
        transpose_cast_kernel<<<(DIMc * DIMc + 255) / 256, 256, 0, stream>>>(wq, wqkT, DIMc, DIMc);
        transpose_cast_kernel<<<(DIMc * DIMc + 255) / 256, 256, 0, stream>>>(wv, wvT,  DIMc, DIMc);
        transpose_cast_kernel<<<(DIMc * DIMc + 255) / 256, 256, 0, stream>>>(wo, woT,  DIMc, DIMc);
        transpose_cast_kernel<<<(DIMc * FF   + 255) / 256, 256, 0, stream>>>(w1, wf1T, DIMc, FF);
        transpose_cast_kernel<<<(FF * DIMc   + 255) / 256, 256, 0, stream>>>(w2, wf2T, FF, DIMc);

        // --- attention half: x1 += attn(LN(x2)) ---
        ln_kernel<<<Rc, 32, 0, stream>>>(x2, ln1_s + l * DIMc, ln1_b + l * DIMc, xn);
        gemm_kernel<false, false, false, true, false>
            <<<dim3(Rc / 16, DIMc / 256), gblk, 0, stream>>>(xn, wqkT, nullptr, qk, nullptr, DIMc, DIMc);
        gemm_kernel<false, false, false, true, false>
            <<<dim3(Rc / 16, DIMc / 256), gblk, 0, stream>>>(xn, wvT, nullptr, vv, nullptr, DIMc, DIMc);
        bucket_kernel<<<(BH * NHASH * Tc + 255) / 256, 256, 0, stream>>>(
            qk, rot + (size_t)l * DH * NHASH * (NBUCK / 2), bkt);
        sort_kernel<<<BH, 256, 0, stream>>>(bkt, st, undo);
        gather_kernel<<<(BH * Sc + 255) / 256, 256, 0, stream>>>(qk, vv, st, sqk, bkn, svb);
        attn_kernel<<<BH * CHUNKS, gblk, 0, stream>>>(sqk, bkn, svb, st, sob, slog);
        combine_kernel<<<(BH * Tc + 255) / 256, 256, 0, stream>>>(undo, slog, sob, attn);
        gemm_kernel<true, false, true, true, false>
            <<<dim3(Rc / 16, DIMc / 256), gblk, 0, stream>>>(attn, woT, b_out + l * DIMc, x1, nullptr, DIMc, DIMc);

        // --- feedforward half: x2 += FF(LN(x1)) ---
        ln_kernel<<<Rc, 32, 0, stream>>>(x1, ln2_s + l * DIMc, ln2_b + l * DIMc, xn);
        gemm_kernel<true, true, false, false, true>
            <<<dim3(Rc / 16, FF / 256), gblk, 0, stream>>>(xn, wf1T, b_ff1 + l * FF, nullptr, hid, DIMc, FF);
        gemm_kernel<true, false, true, true, false>
            <<<dim3(Rc / 16, DIMc / 256), gblk, 0, stream>>>(hid, wf2T, b_ff2 + l * DIMc, x2, nullptr, FF, DIMc);
    }

    cls_kernel<<<1, 32, 0, stream>>>(x1, x2, w_cls, b_cls, out);
}
